// FeaturePropogation_5231270166851
// MI455X (gfx1250) — matrix operand order repaired and checked by reading, not compiled, not run on hardware
//
#include <hip/hip_runtime.h>
#include <hip/hip_bf16.h>
#include <stdint.h>

typedef __attribute__((ext_vector_type(16))) __bf16 v16bf;
typedef __attribute__((ext_vector_type(8)))  __bf16 v8bf;
typedef __attribute__((ext_vector_type(8)))  float  v8f;
typedef __attribute__((ext_vector_type(4)))  unsigned int v4u;
typedef __attribute__((ext_vector_type(4)))  int    v4i;
typedef __attribute__((ext_vector_type(8)))  int    v8i;

#define B_        4
#define N1_       16384
#define N2_       4096
#define C1_       64
#define C2_       256
#define CM_       256
#define CO_       256
#define K1_       320            // C1+C2
#define NCOL_     65536          // B*N1 (GEMM N dimension)
#define SLICES_   32
#define SLICE_LEN_ 2048          // NCOL_/SLICES_

// ---------------------------------------------------------------------------
// f32 -> bf16 conversion (for the small weight matrices)
// ---------------------------------------------------------------------------
__global__ __launch_bounds__(256) void cvt_bf16_kernel(const float* __restrict__ src,
                                                       __bf16* __restrict__ dst, int n) {
  int i = blockIdx.x * 256 + threadIdx.x;
  if (i < n) dst[i] = (__bf16)src[i];
}

// ---------------------------------------------------------------------------
// Transpose f2 [B][C2][N2] -> f2t [B][N2][C2] so gathers read coalesced rows
// ---------------------------------------------------------------------------
__global__ __launch_bounds__(256) void transpose_f2_kernel(const float* __restrict__ f2,
                                                           float* __restrict__ f2t) {
  __shared__ float tile[32][33];
  const int b  = blockIdx.z;
  const int c0 = blockIdx.y * 32;
  const int n0 = blockIdx.x * 32;
  const int tx = threadIdx.x & 31;
  const int ty = threadIdx.x >> 5;      // 0..7
#pragma unroll
  for (int i = 0; i < 32; i += 8)
    tile[ty + i][tx] = f2[((size_t)b * C2_ + (c0 + ty + i)) * N2_ + (n0 + tx)];
  __syncthreads();
#pragma unroll
  for (int i = 0; i < 32; i += 8)
    f2t[((size_t)b * N2_ + (n0 + ty + i)) * C2_ + (c0 + tx)] = tile[tx][ty + i];
}

// ---------------------------------------------------------------------------
// Brute-force 3-NN + inverse-distance weights. p2 tiles staged through LDS.
// ---------------------------------------------------------------------------
__global__ __launch_bounds__(256) void knn_kernel(const float* __restrict__ p1,
                                                  const float* __restrict__ p2,
                                                  int* __restrict__ idx,
                                                  float* __restrict__ wts) {
  constexpr int TP = 1024;
  __shared__ float sx[TP], sy[TP], sz[TP];
  const int blocksPerBatch = N1_ / 256;            // 64
  const int b = blockIdx.x / blocksPerBatch;
  const int n = (blockIdx.x % blocksPerBatch) * 256 + threadIdx.x;

  const float* q = &p1[((size_t)b * N1_ + n) * 3];
  const float qx = q[0], qy = q[1], qz = q[2];

  float d0 = 1e30f, d1 = 1e30f, d2 = 1e30f;
  int   i0 = 0,     i1 = 0,     i2 = 0;

  for (int t0 = 0; t0 < N2_; t0 += TP) {
    __syncthreads();
    for (int i = threadIdx.x; i < TP; i += 256) {
      const float* p = &p2[((size_t)b * N2_ + t0 + i) * 3];
      sx[i] = p[0]; sy[i] = p[1]; sz[i] = p[2];
    }
    __syncthreads();
    for (int i = 0; i < TP; ++i) {
      const float dx = qx - sx[i], dy = qy - sy[i], dz = qz - sz[i];
      const float d = dx * dx + dy * dy + dz * dz;
      const int gi = t0 + i;
      if (d < d2) {
        if (d < d1) {
          if (d < d0) { d2 = d1; i2 = i1; d1 = d0; i1 = i0; d0 = d; i0 = gi; }
          else        { d2 = d1; i2 = i1; d1 = d;  i1 = gi; }
        } else        { d2 = d;  i2 = gi; }
      }
    }
  }
  const float w0 = 1.f / (d0 + 1e-8f);
  const float w1 = 1.f / (d1 + 1e-8f);
  const float w2 = 1.f / (d2 + 1e-8f);
  const float sinv = 1.f / (w0 + w1 + w2);
  const size_t o = ((size_t)b * N1_ + n) * 3;
  idx[o] = i0; idx[o + 1] = i1; idx[o + 2] = i2;
  wts[o] = w0 * sinv; wts[o + 1] = w1 * sinv; wts[o + 2] = w2 * sinv;
}

// ---------------------------------------------------------------------------
// Build GEMM1 B-matrix: Xc [j=0..65535][k=0..319] bf16 (K contiguous per column)
// ---------------------------------------------------------------------------
__global__ __launch_bounds__(320) void build_x_kernel(const float* __restrict__ f1,
                                                      const float* __restrict__ f2t,
                                                      const int* __restrict__ idx,
                                                      const float* __restrict__ wts,
                                                      __bf16* __restrict__ Xc) {
  const int j = blockIdx.x;
  const int b = j >> 14;
  const int n = j & (N1_ - 1);
  const int t = threadIdx.x;
  const size_t o3 = (size_t)j * 3;
  const int i0 = idx[o3], i1 = idx[o3 + 1], i2 = idx[o3 + 2];
  const float w0 = wts[o3], w1 = wts[o3 + 1], w2 = wts[o3 + 2];
  float v;
  if (t < C1_) {
    v = f1[((size_t)b * C1_ + t) * N1_ + n];
  } else {
    const int c = t - C1_;
    const float* base = f2t + (size_t)b * N2_ * C2_;
    v = w0 * base[(size_t)i0 * C2_ + c]
      + w1 * base[(size_t)i1 * C2_ + c]
      + w2 * base[(size_t)i2 * C2_ + c];
  }
  Xc[(size_t)j * K1_ + t] = (__bf16)v;
}

// ---------------------------------------------------------------------------
// TDM: DMA a 2D tile (tile0 x tile1, 2-byte elements, row stride = stride0
// elements) from global memory into LDS, inserting 4 DWORDs (8 bf16) of
// padding after every 32 DWORDs (64 bf16 = one tile row) -> LDS rows of 72.
// D# packed per CDNA5 ISA 8.3/8.4. This toolchain exposes the 6-arg builtin.
// ---------------------------------------------------------------------------
__device__ __forceinline__ void tdm_load_2d(const void* gptr, uint32_t lds_off,
                                            uint32_t dim0, uint32_t dim1,
                                            uint32_t tile0, uint32_t tile1,
                                            uint32_t stride0) {
  const uint64_t ga = (uint64_t)(uintptr_t)gptr;
  v4u g0;
  g0[0] = 1u;                                                 // count=1, user load
  g0[1] = lds_off;                                            // lds_addr (bytes)
  g0[2] = (uint32_t)ga;                                       // global_addr[31:0]
  g0[3] = ((uint32_t)(ga >> 32) & 0x01FFFFFFu) | (2u << 30);  // addr[56:32] | type=2
  v8i g1;
  g1[0] = (int)((1u << 16) |            // data_size = 2 bytes
                (1u << 20) |            // pad_enable
                (4u << 22) |            // pad_interval: 32 DWORDs
                (3u << 25));            // pad_amount:   4 DWORDs
  g1[1] = (int)((dim0 & 0xFFFFu) << 16);                        // tensor_dim0 lo
  g1[2] = (int)((dim0 >> 16) | ((dim1 & 0xFFFFu) << 16));       // dim0 hi | dim1 lo
  g1[3] = (int)((dim1 >> 16) | (tile0 << 16));                  // dim1 hi | tile_dim0
  g1[4] = (int)tile1;                                           // tile_dim1 (tile_dim2=0)
  g1[5] = (int)stride0;                                         // tensor_dim0_stride lo
  g1[6] = 0;                                                    // stride0 hi | stride1 lo
  g1[7] = 0;                                                    // stride1 hi
  v4i z4 = {0, 0, 0, 0};
  v8i z8 = {0, 0, 0, 0, 0, 0, 0, 0};
  __builtin_amdgcn_tensor_load_to_lds(g0, g1, z4, z4, z8, 0);
}

// ---------------------------------------------------------------------------
// WMMA fragment loaders (LDS tile stored [row][72] bf16, 16B-aligned chunks)
// A 16x32 bf16: lane<16 -> row=lane, K 0-15 ; lane>=16 -> row=lane-16, K 16-31
// B 32x16 bf16: lane<16 -> col=lane, K 0-15 ; lane>=16 -> col=lane-16, K 16-31
// ---------------------------------------------------------------------------
__device__ __forceinline__ v16bf load_a_frag(const __bf16* p) {
  v8bf lo = *(const v8bf*)p;
  v8bf hi = *(const v8bf*)(p + 8);
  return __builtin_shufflevector(lo, hi, 0,1,2,3,4,5,6,7,8,9,10,11,12,13,14,15);
}
__device__ __forceinline__ v16bf load_b_frag(const __bf16* p) {
  v8bf lo = *(const v8bf*)p;
  v8bf hi = *(const v8bf*)(p + 8);
  return __builtin_shufflevector(lo, hi, 0,1,2,3,4,5,6,7,8,9,10,11,12,13,14,15);
}

// ---------------------------------------------------------------------------
// GEMM1: Y1[256][65536] (bf16) = W1b[256][320] x Xc[320][65536]
// TDM double-buffered tile staging; 8 waves; block tile 128x128.
// ---------------------------------------------------------------------------
__global__ __launch_bounds__(256) void gemm1_kernel(const __bf16* __restrict__ A,
                                                    const __bf16* __restrict__ Bm,
                                                    __bf16* __restrict__ Y) {
  constexpr int K = K1_;
  constexpr int NT = K / 64;                 // 5 K-tiles of 64
  __shared__ __bf16 lA[2][128][72];
  __shared__ __bf16 lB[2][128][72];
  const int tid = threadIdx.x;
  const int lane = tid & 31;
  const int wv = tid >> 5;
  const int waveM = wv & 3, waveN = wv >> 2;
  const int m0 = blockIdx.y * 128;
  const int n0 = blockIdx.x * 128;
  const int half = lane >> 4;
  const int l16 = lane & 15;
  const bool issuer = (wv == 0);

  v8f acc[2][4] = {};

  if (issuer) {
    tdm_load_2d(A  + (size_t)m0 * K, (uint32_t)(uintptr_t)&lA[0][0][0], K, 128, 64, 128, K);
    tdm_load_2d(Bm + (size_t)n0 * K, (uint32_t)(uintptr_t)&lB[0][0][0], K, 128, 64, 128, K);
  }

  int p = 0;
  for (int t = 0; t < NT; ++t) {
    __builtin_amdgcn_s_wait_tensorcnt(0);   // no-op for waves with TENSORcnt==0
    __syncthreads();                        // tile t visible; buffers p^1 free
    if (issuer && (t + 1 < NT)) {
      const int k0 = (t + 1) * 64;
      tdm_load_2d(A  + (size_t)m0 * K + k0, (uint32_t)(uintptr_t)&lA[p ^ 1][0][0],
                  K, 128, 64, 128, K);
      tdm_load_2d(Bm + (size_t)n0 * K + k0, (uint32_t)(uintptr_t)&lB[p ^ 1][0][0],
                  K, 128, 64, 128, K);
    }
#pragma unroll
    for (int kt = 0; kt < 64; kt += 32) {
      v16bf a[2], bfr[4];
      const int rowA = waveM * 32 + l16;
      const int colB = waveN * 64 + l16;
#pragma unroll
      for (int fm = 0; fm < 2; ++fm)
        a[fm] = load_a_frag(&lA[p][rowA + fm * 16][kt + half * 16]);
#pragma unroll
      for (int fn = 0; fn < 4; ++fn)
        bfr[fn] = load_b_frag(&lB[p][colB + fn * 16][kt + half * 16]);
#pragma unroll
      for (int fm = 0; fm < 2; ++fm)
#pragma unroll
        for (int fn = 0; fn < 4; ++fn)
          acc[fm][fn] = __builtin_amdgcn_wmma_f32_16x16x32_bf16(
              false, a[fm], false, bfr[fn], (short)0, acc[fm][fn], false, false);
    }
    p ^= 1;
  }

  const int oBase = m0 + waveM * 32 + half * 8;
  const int jBase = n0 + waveN * 64 + l16;
#pragma unroll
  for (int fm = 0; fm < 2; ++fm)
#pragma unroll
    for (int fn = 0; fn < 4; ++fn)
#pragma unroll
      for (int r = 0; r < 8; ++r) {
        const int o = oBase + fm * 16 + r;
        const int j = jBase + fn * 16;
        Y[(size_t)o * NCOL_ + j] = (__bf16)acc[fm][fn][r];
      }
}

// ---------------------------------------------------------------------------
// GEMM2: out(raw) = W2b[256][256] x relu(bn1(Y1))[256][65536]
// BN1+ReLU fused into B-tile staging (transpose Y1[k][j] -> LDS[j][k]),
// so TDM is not applicable here; manual staging kept.
// ---------------------------------------------------------------------------
__global__ __launch_bounds__(256) void gemm2_kernel(const __bf16* __restrict__ A,
                                                    const __bf16* __restrict__ Y1,
                                                    const float* __restrict__ scale1,
                                                    const float* __restrict__ shift1,
                                                    float* __restrict__ Out) {
  constexpr int K = CM_;
  __shared__ __bf16 lA[128][72];
  __shared__ __bf16 lB[128][72];
  const int tid = threadIdx.x;
  const int lane = tid & 31;
  const int wv = tid >> 5;
  const int waveM = wv & 3, waveN = wv >> 2;
  const int m0 = blockIdx.y * 128;
  const int n0 = blockIdx.x * 128;
  const int half = lane >> 4;
  const int l16 = lane & 15;

  v8f acc[2][4] = {};

  for (int k0 = 0; k0 < K; k0 += 64) {
    for (int i = tid; i < 128 * 32; i += 256) {
      const int r = i >> 5;
      const int c = (i & 31) << 1;
      *(uint32_t*)&lA[r][c] = *(const uint32_t*)&A[(size_t)(m0 + r) * K + (k0 + c)];
    }
    for (int i = tid; i < 64 * 64; i += 256) {
      const int k = i >> 6;               // 0..63
      const int jp = (i & 63) << 1;       // 0..126 step 2
      const int kg = k0 + k;
      union { uint32_t u; __bf16 h[2]; } cv;
      cv.u = *(const uint32_t*)&Y1[(size_t)kg * NCOL_ + (n0 + jp)];
      const float sc = scale1[kg], sh = shift1[kg];
      float v0 = (float)cv.h[0] * sc + sh; v0 = v0 > 0.f ? v0 : 0.f;
      float v1 = (float)cv.h[1] * sc + sh; v1 = v1 > 0.f ? v1 : 0.f;
      lB[jp][k]     = (__bf16)v0;
      lB[jp + 1][k] = (__bf16)v1;
    }
    __syncthreads();
#pragma unroll
    for (int kt = 0; kt < 64; kt += 32) {
      v16bf a[2], bfr[4];
      const int rowA = waveM * 32 + l16;
      const int colB = waveN * 64 + l16;
#pragma unroll
      for (int fm = 0; fm < 2; ++fm)
        a[fm] = load_a_frag(&lA[rowA + fm * 16][kt + half * 16]);
#pragma unroll
      for (int fn = 0; fn < 4; ++fn)
        bfr[fn] = load_b_frag(&lB[colB + fn * 16][kt + half * 16]);
#pragma unroll
      for (int fm = 0; fm < 2; ++fm)
#pragma unroll
        for (int fn = 0; fn < 4; ++fn)
          acc[fm][fn] = __builtin_amdgcn_wmma_f32_16x16x32_bf16(
              false, a[fm], false, bfr[fn], (short)0, acc[fm][fn], false, false);
    }
    __syncthreads();
  }

  const int oBase = m0 + waveM * 32 + half * 8;
  const int jBase = n0 + waveN * 64 + l16;
#pragma unroll
  for (int fm = 0; fm < 2; ++fm)
#pragma unroll
    for (int fn = 0; fn < 4; ++fn)
#pragma unroll
      for (int r = 0; r < 8; ++r) {
        const int o = oBase + fm * 16 + r;
        const int j = jBase + fn * 16;
        const int bb = j >> 14, nn = j & (N1_ - 1);
        Out[(size_t)bb * (CO_ * N1_) + (size_t)o * N1_ + nn] = acc[fm][fn][r];
      }
}

// ---------------------------------------------------------------------------
// Deterministic per-channel partial sums (sum, sumsq) -> parts[o][slice]
// ---------------------------------------------------------------------------
__global__ __launch_bounds__(256) void stats_y1_kernel(const __bf16* __restrict__ Y,
                                                       float* __restrict__ ps,
                                                       float* __restrict__ pq) {
  const int o = blockIdx.y, s = blockIdx.x, tid = threadIdx.x;
  const size_t base = (size_t)o * NCOL_ + (size_t)s * SLICE_LEN_;
  float sm = 0.f, sq = 0.f;
  for (int i = tid; i < SLICE_LEN_; i += 256) {
    const float v = (float)Y[base + i];
    sm += v; sq += v * v;
  }
  __shared__ float ls[256], lq[256];
  ls[tid] = sm; lq[tid] = sq; __syncthreads();
  for (int st = 128; st > 0; st >>= 1) {
    if (tid < st) { ls[tid] += ls[tid + st]; lq[tid] += lq[tid + st]; }
    __syncthreads();
  }
  if (tid == 0) { ps[o * SLICES_ + s] = ls[0]; pq[o * SLICES_ + s] = lq[0]; }
}

__global__ __launch_bounds__(256) void stats_out_kernel(const float* __restrict__ Out,
                                                        float* __restrict__ ps,
                                                        float* __restrict__ pq) {
  const int o = blockIdx.y, s = blockIdx.x, tid = threadIdx.x;
  float sm = 0.f, sq = 0.f;
  for (int i = tid; i < SLICE_LEN_; i += 256) {
    const int m = s * SLICE_LEN_ + i;                 // flat (b,n) index
    const int bb = m >> 14, nn = m & (N1_ - 1);
    const float v = Out[(size_t)bb * (CO_ * N1_) + (size_t)o * N1_ + nn];
    sm += v; sq += v * v;
  }
  __shared__ float ls[256], lq[256];
  ls[tid] = sm; lq[tid] = sq; __syncthreads();
  for (int st = 128; st > 0; st >>= 1) {
    if (tid < st) { ls[tid] += ls[tid + st]; lq[tid] += lq[tid + st]; }
    __syncthreads();
  }
  if (tid == 0) { ps[o * SLICES_ + s] = ls[0]; pq[o * SLICES_ + s] = lq[0]; }
}

// ---------------------------------------------------------------------------
// Fold partials into BN affine params: scale = g*rsqrt(var+eps), shift = b-mu*scale
// ---------------------------------------------------------------------------
__global__ void bn_params_kernel(const float* __restrict__ ps, const float* __restrict__ pq,
                                 const float* __restrict__ g, const float* __restrict__ bta,
                                 float* __restrict__ scale, float* __restrict__ shift) {
  const int o = threadIdx.x;                 // 256 threads, 1 block
  float sm = 0.f, sq = 0.f;
  for (int s = 0; s < SLICES_; ++s) { sm += ps[o * SLICES_ + s]; sq += pq[o * SLICES_ + s]; }
  const float inv = 1.0f / (float)NCOL_;
  const float mu = sm * inv;
  const float var = sq * inv - mu * mu;
  const float rstd = rsqrtf(var + 1e-5f);
  const float sc = rstd * g[o];
  scale[o] = sc;
  shift[o] = bta[o] - mu * sc;
}

// ---------------------------------------------------------------------------
// In-place BN2 + ReLU on d_out ([b][o][n] layout)
// ---------------------------------------------------------------------------
__global__ __launch_bounds__(256) void finalize_kernel(float* __restrict__ out,
                                                       const float* __restrict__ scale,
                                                       const float* __restrict__ shift) {
  const size_t i = (size_t)blockIdx.x * 256 + threadIdx.x;
  const int o = (int)((i >> 14) & 255);
  const float v = out[i] * scale[o] + shift[o];
  out[i] = v > 0.f ? v : 0.f;
}

// ---------------------------------------------------------------------------
extern "C" void kernel_launch(void* const* d_in, const int* in_sizes, int n_in,
                              void* d_out, int out_size, void* d_ws, size_t ws_size,
                              hipStream_t stream) {
  const float* p1 = (const float*)d_in[0];
  const float* p2 = (const float*)d_in[1];
  const float* f1 = (const float*)d_in[2];
  const float* f2 = (const float*)d_in[3];
  const float* W1 = (const float*)d_in[4];
  const float* g1 = (const float*)d_in[5];
  const float* b1 = (const float*)d_in[6];
  const float* W2 = (const float*)d_in[7];
  const float* g2 = (const float*)d_in[8];
  const float* b2 = (const float*)d_in[9];
  float* out = (float*)d_out;

  char* ws = (char*)d_ws;
  size_t off = 0;
  auto carve = [&](size_t bytes) -> char* {
    char* p = ws + off;
    off += (bytes + 255) & ~(size_t)255;
    return p;
  };
  int*    idx   = (int*)   carve((size_t)NCOL_ * 3 * sizeof(int));
  float*  wts   = (float*) carve((size_t)NCOL_ * 3 * sizeof(float));
  float*  f2t   = (float*) carve((size_t)B_ * N2_ * C2_ * sizeof(float));
  __bf16* Xc    = (__bf16*)carve((size_t)NCOL_ * K1_ * sizeof(__bf16));
  __bf16* W1b   = (__bf16*)carve((size_t)CM_ * K1_ * sizeof(__bf16));
  __bf16* W2b   = (__bf16*)carve((size_t)CO_ * CM_ * sizeof(__bf16));
  __bf16* Y1    = (__bf16*)carve((size_t)CM_ * NCOL_ * sizeof(__bf16));
  float*  ps1   = (float*) carve((size_t)CM_ * SLICES_ * sizeof(float));
  float*  pq1   = (float*) carve((size_t)CM_ * SLICES_ * sizeof(float));
  float*  ps2   = (float*) carve((size_t)CO_ * SLICES_ * sizeof(float));
  float*  pq2   = (float*) carve((size_t)CO_ * SLICES_ * sizeof(float));
  float*  sc1   = (float*) carve(CM_ * sizeof(float));
  float*  sh1   = (float*) carve(CM_ * sizeof(float));
  float*  sc2   = (float*) carve(CO_ * sizeof(float));
  float*  sh2   = (float*) carve(CO_ * sizeof(float));
  (void)in_sizes; (void)n_in; (void)out_size; (void)ws_size;

  // weights -> bf16
  cvt_bf16_kernel<<<(CM_ * K1_ + 255) / 256, 256, 0, stream>>>(W1, W1b, CM_ * K1_);
  cvt_bf16_kernel<<<(CO_ * CM_ + 255) / 256, 256, 0, stream>>>(W2, W2b, CO_ * CM_);

  // f2 transpose for coalesced gathers
  transpose_f2_kernel<<<dim3(N2_ / 32, C2_ / 32, B_), 256, 0, stream>>>(f2, f2t);

  // 3-NN + weights
  knn_kernel<<<(B_ * N1_) / 256, 256, 0, stream>>>(p1, p2, idx, wts);

  // interpolate + concat -> Xc (bf16, K-contiguous columns)
  build_x_kernel<<<NCOL_, K1_, 0, stream>>>(f1, f2t, idx, wts, Xc);

  // conv1 as WMMA GEMM with TDM double-buffered staging
  gemm1_kernel<<<dim3(NCOL_ / 128, CM_ / 128), 256, 0, stream>>>(W1b, Xc, Y1);

  // BN1 batch statistics (deterministic two-pass)
  stats_y1_kernel<<<dim3(SLICES_, CM_), 256, 0, stream>>>(Y1, ps1, pq1);
  bn_params_kernel<<<1, CM_, 0, stream>>>(ps1, pq1, g1, b1, sc1, sh1);

  // conv2 as WMMA GEMM with fused BN1+ReLU on the input tiles
  gemm2_kernel<<<dim3(NCOL_ / 128, CO_ / 128), 256, 0, stream>>>(W2b, Y1, sc1, sh1, out);

  // BN2 statistics + in-place BN2+ReLU on d_out
  stats_out_kernel<<<dim3(SLICES_, CO_), 256, 0, stream>>>(out, ps2, pq2);
  bn_params_kernel<<<1, CO_, 0, stream>>>(ps2, pq2, g2, b2, sc2, sh2);
  finalize_kernel<<<(B_ * CO_ * N1_) / 256, 256, 0, stream>>>(out, sc2, sh2);
}
